// GATResMeanConv_31980326486698
// MI455X (gfx1250) — compile-verified, matrix operand order checked by reading, not run on hardware
//
#include <hip/hip_runtime.h>

#define DHID 256
#define DEDGE 32
#define LBLK 5
#define NEG_SLOPE 0.2f
#define EPSF 1e-16f

typedef __attribute__((ext_vector_type(8)))  __bf16 v8bf;
typedef __attribute__((ext_vector_type(16))) __bf16 v16bf;
typedef __attribute__((ext_vector_type(8)))  float  v8f;

__device__ __forceinline__ unsigned short f2bf(float f) {
  unsigned u = __float_as_uint(f);
  u += 0x7FFFu + ((u >> 16) & 1u);          // round-to-nearest-even
  return (unsigned short)(u >> 16);
}
__device__ __forceinline__ float bf2f(unsigned short h) {
  return __uint_as_float(((unsigned)h) << 16);
}
// order-preserving float<->uint encoding for atomicMax on floats
__device__ __forceinline__ unsigned encf(float f) {
  unsigned u = __float_as_uint(f);
  return (u & 0x80000000u) ? ~u : (u | 0x80000000u);
}
__device__ __forceinline__ float decf(unsigned e) {
  unsigned u = (e & 0x80000000u) ? (e & 0x7FFFFFFFu) : ~e;
  return __uint_as_float(u);
}

// ---------------------------------------------------------------- fill
__global__ void fill_u32_k(unsigned* __restrict__ p, unsigned v, int n) {
  int i = blockIdx.x * blockDim.x + threadIdx.x;
  if (i < n) p[i] = v;
}

// ---------------------------------------------------------------- degree
__global__ void deg_k(const int* __restrict__ dst, float* __restrict__ deg, int E) {
  int e = blockIdx.x * blockDim.x + threadIdx.x;
  if (e < E) atomicAdd(&deg[dst[e]], 1.0f);
}

// ---------------------------------------------------------------- lin0: h = x@W0 + b0 (DIN=3)
__global__ void lin0_k(const float* __restrict__ x, const float* __restrict__ W0,
                       const float* __restrict__ b0, float* __restrict__ h,
                       unsigned short* __restrict__ hbf, int N) {
  int n = blockIdx.x;
  int d = threadIdx.x;
  if (n >= N) return;
  float v = b0[d] + x[(size_t)n*3+0]*W0[d] + x[(size_t)n*3+1]*W0[DHID+d] + x[(size_t)n*3+2]*W0[2*DHID+d];
  size_t o = (size_t)n * DHID + d;
  h[o] = v;
  hbf[o] = f2bf(v);
}

// ---------------------------------------------------------------- weight repack (f32 KxN -> bf16 WMMA-B lane layout)
// packed element index: (((ct*8 + kt)*32 + lane)*16 + j)
// lane -> column n = ct*16 + (lane&15);  half lh = lane>>4
// j<8  -> k = kt*32 + lh*8 + j ;  j>=8 -> k = kt*32 + 16 + lh*8 + (j-8)
__global__ void pack_b_k(const float* __restrict__ W, unsigned short* __restrict__ out) {
  int idx = blockIdx.x * blockDim.x + threadIdx.x;   // 0 .. 65535
  int j    = idx & 15;
  int lane = (idx >> 4) & 31;
  int kt   = (idx >> 9) & 7;
  int ct   = idx >> 12;
  int lh = lane >> 4;
  int n  = ct * 16 + (lane & 15);
  int k  = kt * 32 + (j < 8 ? (lh * 8 + j) : (16 + lh * 8 + (j - 8)));
  out[idx] = f2bf(W[(size_t)k * DHID + n]);
}

// ---------------------------------------------------------------- WMMA GEMM: Out = epilogue(A(bf16, MxK row-major) @ Bpacked)
// wave computes a 16x64 tile; 4 waves / block; grid = (ceil(Mtiles/4), DHID/64)
// K-loop software-pipelined: k+1 fragments are issued before k's WMMAs.
template<bool ADD_C, bool BIAS, bool RELU, bool RES, bool WBF>
__global__ void wmma_gemm_k(const unsigned short* __restrict__ A,
                            const unsigned short* __restrict__ Bp,
                            const float* __restrict__ Cin,
                            const float* __restrict__ bias,
                            const float* __restrict__ Res,
                            float* __restrict__ Out,
                            unsigned short* __restrict__ OutBf,
                            int Mtiles, int Nrows) {
  int wave = threadIdx.x >> 5;
  int lane = threadIdx.x & 31;
  int rt = blockIdx.x * 4 + wave;          // 16-row tile
  if (rt >= Mtiles) return;                // wave-uniform: EXEC stays all-ones for live waves
  int m0  = rt * 16;
  int ct0 = blockIdx.y * 4;                // base 16-col tile
  int lh = lane >> 4;
  int ln = lane & 15;
  int mA = m0 + ln; if (mA > Nrows - 1) mA = Nrows - 1;
  const unsigned short* Arow  = A + (size_t)mA * DHID;
  const unsigned short* Bbase = Bp + (((size_t)ct0 * 8) * 32 + lane) * 16;

  auto loadA = [&](int kt) -> v16bf {
    v8bf alo = *(const v8bf*)(Arow + kt * 32 + lh * 8);
    v8bf ahi = *(const v8bf*)(Arow + kt * 32 + 16 + lh * 8);
    return __builtin_shufflevector(alo, ahi, 0,1,2,3,4,5,6,7,8,9,10,11,12,13,14,15);
  };
  auto loadB = [&](int kt, int c) -> v16bf {
    return *(const v16bf*)(Bbase + ((size_t)(c * 8 + kt)) * 32 * 16);
  };

  v8f acc[4] = {v8f{}, v8f{}, v8f{}, v8f{}};
  v16bf avC = loadA(0);
  v16bf bC0 = loadB(0, 0), bC1 = loadB(0, 1), bC2 = loadB(0, 2), bC3 = loadB(0, 3);
  #pragma unroll
  for (int kt = 0; kt < 8; ++kt) {
    v16bf avN = avC, bN0 = bC0, bN1 = bC1, bN2 = bC2, bN3 = bC3;
    if (kt < 7) {                           // issue next-step loads before this step's WMMAs
      avN = loadA(kt + 1);
      bN0 = loadB(kt + 1, 0); bN1 = loadB(kt + 1, 1);
      bN2 = loadB(kt + 1, 2); bN3 = loadB(kt + 1, 3);
    }
    acc[0] = __builtin_amdgcn_wmma_f32_16x16x32_bf16(false, avC, false, bC0, (short)0, acc[0], false, false);
    acc[1] = __builtin_amdgcn_wmma_f32_16x16x32_bf16(false, avC, false, bC1, (short)0, acc[1], false, false);
    acc[2] = __builtin_amdgcn_wmma_f32_16x16x32_bf16(false, avC, false, bC2, (short)0, acc[2], false, false);
    acc[3] = __builtin_amdgcn_wmma_f32_16x16x32_bf16(false, avC, false, bC3, (short)0, acc[3], false, false);
    avC = avN; bC0 = bN0; bC1 = bN1; bC2 = bN2; bC3 = bN3;
  }

  #pragma unroll
  for (int c = 0; c < 4; ++c) {
    int col = (ct0 + c) * 16 + ln;
    float bv = 0.0f;
    if (BIAS) bv = bias[col];
    #pragma unroll
    for (int r = 0; r < 8; ++r) {
      int row = m0 + r + lh * 8;
      if (row >= Nrows) continue;
      size_t o = (size_t)row * DHID + col;
      float v = acc[c][r];
      if (ADD_C) v += Cin[o];
      if (BIAS)  v += bv;
      if (RELU)  v = fmaxf(v, 0.0f);
      if (RES)   v += Res[o];
      Out[o] = v;
      if (WBF) OutBf[o] = f2bf(v);
    }
  }
}

// ---------------------------------------------------------------- per-node dots: s_src = hg.a_src, s_dst = hg.a_dst (wave/node)
__global__ void sdot_k(const float* __restrict__ hg, const float* __restrict__ asrc,
                       const float* __restrict__ adst, float* __restrict__ ssrc,
                       float* __restrict__ sdst, int N) {
  int w = (int)(((size_t)blockIdx.x * blockDim.x + threadIdx.x) >> 5);
  int lane = threadIdx.x & 31;
  if (w >= N) return;
  const float* row = hg + (size_t)w * DHID + lane * 8;
  float s1 = 0.f, s2 = 0.f;
  #pragma unroll
  for (int j = 0; j < 8; ++j) {
    float v = row[j];
    s1 += v * asrc[lane * 8 + j];
    s2 += v * adst[lane * 8 + j];
  }
  for (int off = 16; off; off >>= 1) {
    s1 += __shfl_xor(s1, off, 32);
    s2 += __shfl_xor(s2, off, 32);
  }
  if (lane == 0) { ssrc[w] = s1; sdst[w] = s2; }
}

// ---------------------------------------------------------------- w_ea[k] = sum_d We[k][d]*a_e[d]  (one block of 32)
__global__ void wea_k(const float* __restrict__ We_l, const float* __restrict__ ae_l,
                      float* __restrict__ w_ea) {
  int k = threadIdx.x;
  float s = 0.f;
  for (int d = 0; d < DHID; ++d) s += We_l[(size_t)k * DHID + d] * ae_l[d];
  w_ea[k] = s;
}

// ---------------------------------------------------------------- edge logit + segment max (encoded atomicMax)
__global__ void logit_k(const int* __restrict__ src, const int* __restrict__ dst,
                        const float* __restrict__ ea, const float* __restrict__ w_ea,
                        const float* __restrict__ ssrc, const float* __restrict__ sdst,
                        float* __restrict__ logit, unsigned* __restrict__ mmax, int E) {
  int e = blockIdx.x * blockDim.x + threadIdx.x;
  if (e >= E) return;
  int d = dst[e];
  float acc = ssrc[src[e]] + sdst[d];
  const float* p = ea + (size_t)e * DEDGE;
  #pragma unroll
  for (int k = 0; k < DEDGE; ++k) acc += p[k] * w_ea[k];
  acc = acc > 0.f ? acc : NEG_SLOPE * acc;
  logit[e] = acc;
  atomicMax(&mmax[d], encf(acc));
}

// ---------------------------------------------------------------- exp + segment sum
__global__ void exp_k(const int* __restrict__ dst, const float* __restrict__ logit,
                      const unsigned* __restrict__ mmax, float* __restrict__ exb,
                      float* __restrict__ den, int E) {
  int e = blockIdx.x * blockDim.x + threadIdx.x;
  if (e >= E) return;
  int d = dst[e];
  float ex = __expf(logit[e] - decf(mmax[d]));
  exb[e] = ex;
  atomicAdd(&den[d], ex);
}

// ---------------------------------------------------------------- edge scatter: out[dst] += a * val_bf16[src]  (wave/edge, 8 f/lane)
template<bool USE_ALPHA>
__global__ void scatter_k(const int* __restrict__ src, const int* __restrict__ dst,
                          const float* __restrict__ exb, const float* __restrict__ den,
                          const unsigned short* __restrict__ valbf,
                          float* __restrict__ out, int E) {
  int e = (int)(((size_t)blockIdx.x * blockDim.x + threadIdx.x) >> 5);
  int lane = threadIdx.x & 31;
  if (e >= E) return;
  int s = src[e], d = dst[e];
  float a = 1.0f;
  if (USE_ALPHA) a = exb[e] / (den[d] + EPSF);
  float* o = out + (size_t)d * DHID + lane * 8;
  uint4 raw = *(const uint4*)(valbf + (size_t)s * DHID + lane * 8);
  atomicAdd(o + 0, a * bf2f((unsigned short)(raw.x & 0xFFFFu)));
  atomicAdd(o + 1, a * bf2f((unsigned short)(raw.x >> 16)));
  atomicAdd(o + 2, a * bf2f((unsigned short)(raw.y & 0xFFFFu)));
  atomicAdd(o + 3, a * bf2f((unsigned short)(raw.y >> 16)));
  atomicAdd(o + 4, a * bf2f((unsigned short)(raw.z & 0xFFFFu)));
  atomicAdd(o + 5, a * bf2f((unsigned short)(raw.z >> 16)));
  atomicAdd(o + 6, a * bf2f((unsigned short)(raw.w & 0xFFFFu)));
  atomicAdd(o + 7, a * bf2f((unsigned short)(raw.w >> 16)));
}

// ---------------------------------------------------------------- h1 = relu(agg + bg) -> bf16
__global__ void h1_k(const float* __restrict__ agg, const float* __restrict__ bg,
                     unsigned short* __restrict__ h1bf, int total) {
  int i = blockIdx.x * blockDim.x + threadIdx.x;
  if (i >= total) return;
  float v = agg[i] + bg[i & (DHID - 1)];
  h1bf[i] = f2bf(fmaxf(v, 0.0f));
}

// ---------------------------------------------------------------- mean scale: mn_bf = (acc / max(deg,1)) -> bf16
__global__ void mscale_k(const float* __restrict__ acc, const float* __restrict__ deg,
                         unsigned short* __restrict__ mnbf, int total) {
  int i = blockIdx.x * blockDim.x + threadIdx.x;
  if (i >= total) return;
  float dg = fmaxf(deg[i >> 8], 1.0f);
  mnbf[i] = f2bf(acc[i] / dg);
}

// ---------------------------------------------------------------- out = h @ W1 + b1 (DOUT=1), wave/node
__global__ void outdot_k(const float* __restrict__ h, const float* __restrict__ w1,
                         const float* __restrict__ b1, float* __restrict__ out, int N) {
  int w = (int)(((size_t)blockIdx.x * blockDim.x + threadIdx.x) >> 5);
  int lane = threadIdx.x & 31;
  if (w >= N) return;
  const float* row = h + (size_t)w * DHID + lane * 8;
  float s = 0.f;
  #pragma unroll
  for (int j = 0; j < 8; ++j) s += row[j] * w1[lane * 8 + j];
  for (int off = 16; off; off >>= 1) s += __shfl_xor(s, off, 32);
  if (lane == 0) out[w] = s + b1[0];
}

// ================================================================ launch
extern "C" void kernel_launch(void* const* d_in, const int* in_sizes, int n_in,
                              void* d_out, int out_size, void* d_ws, size_t ws_size,
                              hipStream_t stream) {
  const float* x    = (const float*)d_in[0];
  const int*   ei   = (const int*)d_in[1];
  const float* ea   = (const float*)d_in[2];
  const float* W0   = (const float*)d_in[3];
  const float* b0   = (const float*)d_in[4];
  const float* W1   = (const float*)d_in[5];
  const float* b1   = (const float*)d_in[6];
  const float* Wg   = (const float*)d_in[7];
  const float* bg   = (const float*)d_in[8];
  const float* asrc = (const float*)d_in[9];
  const float* adst = (const float*)d_in[10];
  const float* We   = (const float*)d_in[11];
  const float* aeV  = (const float*)d_in[12];
  const float* Wsm  = (const float*)d_in[13];
  const float* Wnm  = (const float*)d_in[14];
  const float* bm   = (const float*)d_in[15];

  const int N = in_sizes[0] / 3;
  const int E = in_sizes[1] / 2;
  const int* srcI = ei;
  const int* dstI = ei + E;
  const size_t Nf = (size_t)N * DHID;

  // ---- workspace carve (256B aligned slots)
  char* base = (char*)d_ws;
  size_t off = 0;
  auto carve = [&](size_t bytes) -> char* {
    char* p = base + off;
    off += (bytes + 255) & ~(size_t)255;
    return p;
  };
  float*          h    = (float*)carve(Nf * 4);
  unsigned short* hbf  = (unsigned short*)carve(Nf * 2);
  float*          hg   = (float*)carve(Nf * 4);   // reused as t = h1@Ws
  unsigned short* hgbf = (unsigned short*)carve(Nf * 2);
  float*          agg  = (float*)carve(Nf * 4);   // reused as mean accumulator
  unsigned short* h1bf = (unsigned short*)carve(Nf * 2);
  unsigned short* mnbf = (unsigned short*)carve(Nf * 2);
  float*          ssrc = (float*)carve((size_t)N * 4);
  float*          sdst = (float*)carve((size_t)N * 4);
  float*          deg  = (float*)carve((size_t)N * 4);
  float*          den  = (float*)carve((size_t)N * 4);
  unsigned*       mmax = (unsigned*)carve((size_t)N * 4);
  float*          lg   = (float*)carve((size_t)E * 4);
  float*          exb  = (float*)carve((size_t)E * 4);
  float*          w_ea = (float*)carve(DEDGE * 4);
  unsigned short* pgW  = (unsigned short*)carve((size_t)DHID * DHID * 2);
  unsigned short* pgS  = (unsigned short*)carve((size_t)DHID * DHID * 2);
  unsigned short* pgN  = (unsigned short*)carve((size_t)DHID * DHID * 2);

  const int Mtiles = (N + 15) / 16;
  dim3 gGemm((Mtiles + 3) / 4, DHID / 64);
  const int TB = 256;
  int gE   = (E + TB - 1) / TB;
  int gEw  = (E + 7) / 8;           // wave-per-edge
  int gNw  = (N + 7) / 8;           // wave-per-node
  int gNE  = ((int)Nf + TB - 1) / TB;
  int gN1  = (N + TB - 1) / TB;

  // ---- degree (once)
  fill_u32_k<<<gN1, TB, 0, stream>>>((unsigned*)deg, 0u, N);
  deg_k<<<gE, TB, 0, stream>>>(dstI, deg, E);

  // ---- h = x@W0 + b0
  lin0_k<<<N, DHID, 0, stream>>>(x, W0, b0, h, hbf, N);

  for (int l = 0; l < LBLK; ++l) {
    const float* Wg_l = Wg  + (size_t)l * DHID * DHID;
    const float* Ws_l = Wsm + (size_t)l * DHID * DHID;
    const float* Wn_l = Wnm + (size_t)l * DHID * DHID;

    // pack weights into WMMA-B layout (bf16)
    pack_b_k<<<256, 256, 0, stream>>>(Wg_l, pgW);
    pack_b_k<<<256, 256, 0, stream>>>(Ws_l, pgS);
    pack_b_k<<<256, 256, 0, stream>>>(Wn_l, pgN);
    wea_k<<<1, 32, 0, stream>>>(We + (size_t)l * DEDGE * DHID, aeV + (size_t)l * DHID, w_ea);

    // hg = h @ Wg   (WMMA) -> f32 (for sdot) + bf16 (for gather)
    wmma_gemm_k<false,false,false,false,true><<<gGemm, 128, 0, stream>>>(
        hbf, pgW, nullptr, nullptr, nullptr, hg, hgbf, Mtiles, N);

    // per-node attention dots
    sdot_k<<<gNw, TB, 0, stream>>>(hg, asrc + (size_t)l * DHID, adst + (size_t)l * DHID, ssrc, sdst, N);

    // softmax state + agg buffer reset
    fill_u32_k<<<gN1, TB, 0, stream>>>(mmax, 0u, N);
    fill_u32_k<<<gN1, TB, 0, stream>>>((unsigned*)den, 0u, N);
    fill_u32_k<<<gNE, TB, 0, stream>>>((unsigned*)agg, 0u, (int)Nf);

    // logits + segment max, then exp + segment sum
    logit_k<<<gE, TB, 0, stream>>>(srcI, dstI, ea, w_ea, ssrc, sdst, lg, mmax, E);
    exp_k<<<gE, TB, 0, stream>>>(dstI, lg, mmax, exb, den, E);

    // agg[dst] += alpha * hg[src]   (bf16 gather, f32 atomic accumulate)
    scatter_k<true><<<gEw, TB, 0, stream>>>(srcI, dstI, exb, den, hgbf, agg, E);

    // h1 = relu(agg + bg) -> bf16
    h1_k<<<gNE, TB, 0, stream>>>(agg, bg + (size_t)l * DHID, h1bf, (int)Nf);

    // mean aggregation (reuse agg buffer)
    fill_u32_k<<<gNE, TB, 0, stream>>>((unsigned*)agg, 0u, (int)Nf);
    scatter_k<false><<<gEw, TB, 0, stream>>>(srcI, dstI, nullptr, nullptr, h1bf, agg, E);
    mscale_k<<<gNE, TB, 0, stream>>>(agg, deg, mnbf, (int)Nf);

    // t = h1 @ Ws  (WMMA, into hg buffer)
    wmma_gemm_k<false,false,false,false,false><<<gGemm, 128, 0, stream>>>(
        h1bf, pgS, nullptr, nullptr, nullptr, hg, nullptr, Mtiles, N);

    // h = relu(mean@Wn + t + bm) + h   (WMMA, fused epilogue, in-place, also bf16 out)
    wmma_gemm_k<true,true,true,true,true><<<gGemm, 128, 0, stream>>>(
        mnbf, pgN, hg, bm + (size_t)l * DHID, h, h, hbf, Mtiles, N);
  }

  // out = h @ W1 + b1
  outdot_k<<<gNw, TB, 0, stream>>>(h, W1, b1, (float*)d_out, N);
}
